// TLAloss_75376676045214
// MI455X (gfx1250) — matrix operand order repaired and checked by reading, not compiled, not run on hardware
//
#include <hip/hip_runtime.h>
#include <math.h>

// ---------------------------------------------------------------------------
// Problem constants (match reference)
// ---------------------------------------------------------------------------
#define BB 8192
#define LL 1024
#define DD 768
#define HH 3072
#define NINF_V (-100.0f)
#define INV_TEMP (1.0f / 0.07f)

// ---------------------------------------------------------------------------
// CDNA5 WMMA types (wave32): 16x16x32 bf16 -> f32
// ---------------------------------------------------------------------------
typedef __attribute__((ext_vector_type(16))) __bf16 v16bf;
typedef __attribute__((ext_vector_type(8)))  float  v8f;

union FragBF {
    v16bf v;
    int4  q[2];
};

__device__ __forceinline__ unsigned short f2bf(float f) {
    unsigned int u = __float_as_uint(f);
    unsigned int r = (u + 0x7FFFu + ((u >> 16) & 1u)) >> 16;
    return (unsigned short)r;
}

// Async global->LDS copy of 16 bytes (CDNA5, ASYNCcnt-tracked).
__device__ __forceinline__ void async_copy_b128(unsigned lds_addr, const void* gptr) {
    unsigned long long ga = (unsigned long long)(size_t)gptr;
    asm volatile("global_load_async_to_lds_b128 %0, %1, off"
                 :: "v"(lds_addr), "v"(ga)
                 : "memory");
}
__device__ __forceinline__ void wait_async0() {
    asm volatile("s_wait_asynccnt 0x0" ::: "memory");
}

// ---------------------------------------------------------------------------
// Tiled transpose + f32 -> bf16 convert:  src [R,C] f32  ->  dst [C,R] bf16
// R, C multiples of 32. 256 threads, 32x32 LDS tile.
// ---------------------------------------------------------------------------
__global__ __launch_bounds__(256)
void cvt_bf16_T_kernel(const float* __restrict__ src, unsigned short* __restrict__ dst,
                       int R, int C) {
    __shared__ float tile[32][33];
    const int bx = blockIdx.x * 32;     // C dimension
    const int by = blockIdx.y * 32;     // R dimension
    const int tx = threadIdx.x & 31;
    const int ty = threadIdx.x >> 5;    // 0..7
#pragma unroll
    for (int i = 0; i < 32; i += 8)
        tile[ty + i][tx] = src[(size_t)(by + ty + i) * C + bx + tx];
    __syncthreads();
#pragma unroll
    for (int i = 0; i < 32; i += 8)
        dst[(size_t)(bx + ty + i) * R + by + tx] = f2bf(tile[tx][ty + i]);
}

// ---------------------------------------------------------------------------
// Row L2-normalize f32 [M, D] -> bf16 [M, D]. One block (256 thr) per row.
// ---------------------------------------------------------------------------
__global__ __launch_bounds__(256)
void rownorm_kernel(const float* __restrict__ src, unsigned short* __restrict__ dst,
                    int D) {
    const int row = blockIdx.x;
    const int t   = threadIdx.x;
    const float* s = src + (size_t)row * D;

    __shared__ float red[256];
    float ss = 0.f;
    for (int i = t; i < D; i += 256) { float x = s[i]; ss += x * x; }
    red[t] = ss; __syncthreads();
    for (int st = 128; st > 0; st >>= 1) {
        if (t < st) red[t] += red[t + st];
        __syncthreads();
    }
    const float inv = 1.0f / fmaxf(sqrtf(red[0]), 1e-12f);

    for (int i = t; i < D; i += 256)
        dst[(size_t)row * D + i] = f2bf(s[i] * inv);
}

// ---------------------------------------------------------------------------
// bf16 WMMA NT-GEMM: C[M,N] = A[M,K] @ BT[N,K]^T (+bias) (+relu)
// Block tile 128x128, TK=32, 256 threads = 8 waves (2 M-waves x 4 N-waves),
// each wave computes a 4x2 grid of 16x16 v_wmma_f32_16x16x32_bf16 tiles.
// Double-buffered LDS with global_load_async_to_lds_b128 prefetch.
// EPI: 0 = bf16 out, bias+relu ; 1 = f32 out, bias ; 2 = f32 out, plain
// Requires M%128==0, N%128==0, K%32==0 (true for all uses here).
// ---------------------------------------------------------------------------
template <int EPI>
__global__ __launch_bounds__(256)
void gemm_wmma_kernel(const unsigned short* __restrict__ A, int lda,
                      const unsigned short* __restrict__ BT, int ldbt,
                      const float* __restrict__ bias,
                      void* __restrict__ Cout, int ldc,
                      int K) {
    __shared__ __align__(16) unsigned short As[2][128][40];   // [m][k], padded
    __shared__ __align__(16) unsigned short Bs[2][128][40];   // [n][k], padded

    const int tid  = threadIdx.x;
    const int lane = tid & 31;
    const int wave = tid >> 5;      // 0..7
    const int wm   = wave >> 2;     // 0..1 -> 64-row strip
    const int wn   = wave & 3;      // 0..3 -> 32-col strip
    const int l15  = lane & 15;
    const int lhi  = lane >> 4;

    const int bm = blockIdx.y * 128;
    const int bn = blockIdx.x * 128;

    v8f acc[4][2];
#pragma unroll
    for (int i = 0; i < 4; ++i)
#pragma unroll
        for (int j = 0; j < 2; ++j) acc[i][j] = v8f{};

    // copy indices: each thread moves 32B of A and 32B of B per K-step
    const int c_row = tid >> 1;            // 0..127
    const int c_off = (tid & 1) * 16;      // 0 / 16 (halves)

    const unsigned short* Aptr = A  + (size_t)(bm + c_row) * lda  + c_off;
    const unsigned short* Bptr = BT + (size_t)(bn + c_row) * ldbt + c_off;
    const unsigned la0 = (unsigned)(size_t)&As[0][c_row][c_off];
    const unsigned la1 = (unsigned)(size_t)&As[1][c_row][c_off];
    const unsigned lb0 = (unsigned)(size_t)&Bs[0][c_row][c_off];
    const unsigned lb1 = (unsigned)(size_t)&Bs[1][c_row][c_off];

    const int nk = K >> 5;

    // prefetch tile 0 into buffer 0
    async_copy_b128(la0,      Aptr);
    async_copy_b128(la0 + 16, Aptr + 8);
    async_copy_b128(lb0,      Bptr);
    async_copy_b128(lb0 + 16, Bptr + 8);

    for (int kt = 0; kt < nk; ++kt) {
        const int cur = kt & 1;

        wait_async0();          // this thread's async writes to buf[cur] done
        __syncthreads();        // everyone's writes to buf[cur] visible

        // prefetch next tile into the other buffer (no WAR: other buffer,
        // and writes to buf[cur] only resume after next iteration's barrier)
        if (kt + 1 < nk) {
            const unsigned sa = cur ? la0 : la1;
            const unsigned sb = cur ? lb0 : lb1;
            const unsigned short* ap = Aptr + ((kt + 1) << 5);
            const unsigned short* bp = Bptr + ((kt + 1) << 5);
            async_copy_b128(sa,      ap);
            async_copy_b128(sa + 16, ap + 8);
            async_copy_b128(sb,      bp);
            async_copy_b128(sb + 16, bp + 8);
        }

        // Fragment loads per ISA 16-bit layouts.
        // A (16x32): lane m=l15, halves 0..7 = K[lhi*8 .. +7], 8..15 = K[16+lhi*8 .. +7]
        FragBF af[4];
#pragma unroll
        for (int i = 0; i < 4; ++i) {
            const int m = wm * 64 + i * 16 + l15;
            af[i].q[0] = *(const int4*)&As[cur][m][lhi * 8];
            af[i].q[1] = *(const int4*)&As[cur][m][16 + lhi * 8];
        }
        // B (32x16): lane n=l15, halves 0..7 = K[lhi*8 .. +7], 8..15 = K[16+lhi*8 .. +7]
        FragBF bfr[2];
#pragma unroll
        for (int j = 0; j < 2; ++j) {
            const int n = wn * 32 + j * 16 + l15;
            bfr[j].q[0] = *(const int4*)&Bs[cur][n][lhi * 8];
            bfr[j].q[1] = *(const int4*)&Bs[cur][n][16 + lhi * 8];
        }

#pragma unroll
        for (int i = 0; i < 4; ++i)
#pragma unroll
            for (int j = 0; j < 2; ++j)
                acc[i][j] = __builtin_amdgcn_wmma_f32_16x16x32_bf16(
                    /*neg_a=*/false, af[i].v, /*neg_b=*/false, bfr[j].v,
                    /*c_mod=*/(short)0, acc[i][j],
                    /*reuse_a=*/false, /*reuse_b=*/false);
    }

    // Epilogue. C/D layout: VGPR r -> M = r + (lane>=16 ? 8 : 0), N = lane&15.
#pragma unroll
    for (int i = 0; i < 4; ++i) {
#pragma unroll
        for (int j = 0; j < 2; ++j) {
            const int col = bn + wn * 32 + j * 16 + l15;
            float bv = 0.f;
            if constexpr (EPI != 2) bv = bias[col];
#pragma unroll
            for (int r = 0; r < 8; ++r) {
                const int row = bm + wm * 64 + i * 16 + r + lhi * 8;
                float x = acc[i][j][r] + bv;
                if constexpr (EPI == 0) {
                    x = fmaxf(x, 0.f);
                    ((unsigned short*)Cout)[(size_t)row * ldc + col] = f2bf(x);
                } else {
                    ((float*)Cout)[(size_t)row * ldc + col] = x;
                }
            }
        }
    }
}

// ---------------------------------------------------------------------------
// Zero the scalar output (harness poisons d_out).
// ---------------------------------------------------------------------------
__global__ void zero_kernel(float* p) {
    if (blockIdx.x == 0 && threadIdx.x == 0) *p = 0.f;
}

// ---------------------------------------------------------------------------
// Per-row loss: one block (256 thr) per row of sim [B=8192, L=1024].
// ---------------------------------------------------------------------------
__global__ __launch_bounds__(256)
void loss_kernel(const float* __restrict__ sim, const int* __restrict__ tgt,
                 float* __restrict__ out) {
    const int b = blockIdx.x;
    const int t = threadIdx.x;
    const float* srow = sim + (size_t)b * LL;
    const int*   trow = tgt + (size_t)b * LL;

    float v[4];
    float sum_pos = 0.f, denom_l = 0.f;
    int   npos_l = 0;
#pragma unroll
    for (int j = 0; j < 4; ++j) {
        const int l = t + j * 256;
        const float s = srow[l];
        if (trow[l] > 0) {
            npos_l++; sum_pos += s; denom_l += __expf(s * INV_TEMP);
            v[j] = NINF_V;
        } else {
            v[j] = s;
        }
    }

    __shared__ float redf[256];
    __shared__ int   redi[256];

    redi[t] = npos_l; __syncthreads();
    for (int st = 128; st > 0; st >>= 1) { if (t < st) redi[t] += redi[t + st]; __syncthreads(); }
    const int npos = redi[0]; __syncthreads();

    redf[t] = sum_pos; __syncthreads();
    for (int st = 128; st > 0; st >>= 1) { if (t < st) redf[t] += redf[t + st]; __syncthreads(); }
    const float sum_pos_all = redf[0]; __syncthreads();

    redf[t] = denom_l; __syncthreads();
    for (int st = 128; st > 0; st >>= 1) { if (t < st) redf[t] += redf[t + st]; __syncthreads(); }
    float denom = redf[0]; __syncthreads();

    // hard negatives: extract top n_pos maxima among non-positives
    for (int it = 0; it < npos; ++it) {
        float m = fmaxf(fmaxf(v[0], v[1]), fmaxf(v[2], v[3]));
        redf[t] = m; __syncthreads();
        for (int st = 128; st > 0; st >>= 1) { if (t < st) redf[t] = fmaxf(redf[t], redf[t + st]); __syncthreads(); }
        const float gmax = redf[0]; __syncthreads();

        int myidx = 0x7fffffff;
#pragma unroll
        for (int j = 0; j < 4; ++j)
            if (v[j] == gmax) { int gi = t + j * 256; myidx = (gi < myidx) ? gi : myidx; }
        redi[t] = myidx; __syncthreads();
        for (int st = 128; st > 0; st >>= 1) { if (t < st) redi[t] = (redi[t + st] < redi[t]) ? redi[t + st] : redi[t]; __syncthreads(); }
        const int gidx = redi[0]; __syncthreads();

        if ((gidx & 255) == t) v[gidx >> 8] = NINF_V;   // remove selected element
        denom += __expf(gmax * INV_TEMP);               // uniform across block
    }

    if (t == 0) {
        const float loss_b = __logf(denom) - (sum_pos_all * INV_TEMP) / (float)npos;
        atomicAdd(out, loss_b * (1.0f / (float)BB));
    }
}

// ---------------------------------------------------------------------------
// Launcher
// ---------------------------------------------------------------------------
extern "C" void kernel_launch(void* const* d_in, const int* in_sizes, int n_in,
                              void* d_out, int out_size, void* d_ws, size_t ws_size,
                              hipStream_t stream) {
    (void)in_sizes; (void)n_in; (void)out_size; (void)ws_size;

    const float* text = (const float*)d_in[0];   // [B, D]
    const float* labl = (const float*)d_in[1];   // [L, D]
    const int*   tgt  = (const int*)d_in[2];     // [B, L]
    const float* W1   = (const float*)d_in[3];   // [D, H]
    const float* b1   = (const float*)d_in[4];   // [H]
    const float* W2   = (const float*)d_in[5];   // [H, D]
    const float* b2   = (const float*)d_in[6];   // [D]
    float*       out  = (float*)d_out;

    // workspace layout (256B aligned slabs)
    char* base = (char*)d_ws;
    size_t off = 0;
    auto alloc = [&](size_t bytes) {
        size_t o = off;
        off += (bytes + 255) & ~(size_t)255;
        return o;
    };
    unsigned short* XnT = (unsigned short*)(base + alloc((size_t)BB * DD * 2)); // norm text bf16
    unsigned short* XnL = (unsigned short*)(base + alloc((size_t)LL * DD * 2)); // norm label bf16
    unsigned short* W1T = (unsigned short*)(base + alloc((size_t)HH * DD * 2)); // W1^T [H,D] bf16
    unsigned short* W2T = (unsigned short*)(base + alloc((size_t)DD * HH * 2)); // W2^T [D,H] bf16
    unsigned short* HT  = (unsigned short*)(base + alloc((size_t)BB * HH * 2)); // relu hidden text
    unsigned short* HL  = (unsigned short*)(base + alloc((size_t)LL * HH * 2)); // relu hidden label
    float*          PT  = (float*)(base + alloc((size_t)BB * DD * 4));          // proj text f32
    float*          PL  = (float*)(base + alloc((size_t)LL * DD * 4));          // proj label f32
    unsigned short* Tn  = (unsigned short*)(base + alloc((size_t)BB * DD * 2)); // norm proj text bf16
    unsigned short* Ln  = (unsigned short*)(base + alloc((size_t)LL * DD * 2)); // norm proj label bf16 [L,D]
    float*          SIM = (float*)(base + alloc((size_t)BB * LL * 4));          // similarity f32

    const dim3 blk(256);

    // 1. weights -> transposed bf16 (so GEMMs are NT: B tiles are [N][K] contiguous)
    cvt_bf16_T_kernel<<<dim3(HH / 32, DD / 32), blk, 0, stream>>>(W1, W1T, DD, HH);
    cvt_bf16_T_kernel<<<dim3(DD / 32, HH / 32), blk, 0, stream>>>(W2, W2T, HH, DD);

    // 2. input row-normalize -> bf16
    rownorm_kernel<<<dim3(BB), blk, 0, stream>>>(text, XnT, DD);
    rownorm_kernel<<<dim3(LL), blk, 0, stream>>>(labl, XnL, DD);

    // 3. GEMM1: relu(Xn @ W1 + b1) -> bf16
    gemm_wmma_kernel<0><<<dim3(HH / 128, BB / 128), blk, 0, stream>>>(XnT, DD, W1T, DD, b1, HT, HH, DD);
    gemm_wmma_kernel<0><<<dim3(HH / 128, LL / 128), blk, 0, stream>>>(XnL, DD, W1T, DD, b1, HL, HH, DD);

    // 4. GEMM2: H @ W2 + b2 -> f32
    gemm_wmma_kernel<1><<<dim3(DD / 128, BB / 128), blk, 0, stream>>>(HT, HH, W2T, HH, b2, PT, DD, HH);
    gemm_wmma_kernel<1><<<dim3(DD / 128, LL / 128), blk, 0, stream>>>(HL, HH, W2T, HH, b2, PL, DD, HH);

    // 5. row-normalize projections (labels kept row-major: already B^T for sim GEMM)
    rownorm_kernel<<<dim3(BB), blk, 0, stream>>>(PT, Tn, DD);
    rownorm_kernel<<<dim3(LL), blk, 0, stream>>>(PL, Ln, DD);

    // 6. sim = Tn @ Ln^T  ([B,D] x [L,D]^T -> f32 [B,L])
    gemm_wmma_kernel<2><<<dim3(LL / 128, BB / 128), blk, 0, stream>>>(Tn, DD, Ln, DD, nullptr, SIM, LL, DD);

    // 7. loss
    zero_kernel<<<dim3(1), dim3(64), 0, stream>>>(out);
    loss_kernel<<<dim3(BB), blk, 0, stream>>>(SIM, tgt, out);
}
